// VQExpert_33938831572994
// MI455X (gfx1250) — matrix operand order repaired and checked by reading, not compiled, not run on hardware
//
#include <hip/hip_runtime.h>
#include <hip/hip_bf16.h>

// ---------- types ----------
typedef __attribute__((ext_vector_type(16))) __bf16 v16bf;
typedef __attribute__((ext_vector_type(8)))  __bf16 v8bf;
typedef __attribute__((ext_vector_type(8)))  float  v8f;
typedef __attribute__((ext_vector_type(4)))  float  v4f;
typedef __attribute__((ext_vector_type(4)))  int    v4i;

// ---------- problem constants ----------
#define TOKENS    32768   // B*S = 8*4096
#define IN_FEAT   2048
#define HIDDEN    512
#define CODE_DIM  32
#define CODE_SIZE 256
#define OUT_FEAT  2048

// ---------- CDNA5 async global->LDS path (guarded; falls back cleanly) ----
#if defined(__HIP_DEVICE_COMPILE__) && \
    __has_builtin(__builtin_amdgcn_global_load_async_to_lds_b128) && \
    __has_builtin(__builtin_amdgcn_s_wait_asynccnt)
#define HAVE_ASYNC_LDS 1
#else
#define HAVE_ASYNC_LDS 0
#endif

#define AS1 __attribute__((address_space(1)))
#define AS3 __attribute__((address_space(3)))

template<int NCNT>
__device__ __forceinline__ void wait_asynccnt() {
#if HAVE_ASYNC_LDS
  __builtin_amdgcn_s_wait_asynccnt(NCNT);
#endif
}

__device__ __forceinline__ void async_cp_b128(const void* g, void* l) {
#if HAVE_ASYNC_LDS
  // signature (probe-derived): (AS1 v4i* src, AS3 v4i* dst, imm offset, imm cpol)
  __builtin_amdgcn_global_load_async_to_lds_b128((AS1 v4i*)g, (AS3 v4i*)l, 0, 0);
#else
  (void)g; (void)l;
#endif
}

// =====================================================================
// f32 -> bf16 weight pre-conversion (one shot, ~4 MB total; makes the
// B operand DMA-able by the async-LDS path)
// =====================================================================
__global__ __launch_bounds__(256)
void cvt_f32_to_bf16(const float* __restrict__ in, __bf16* __restrict__ out, int n8)
{
  const int i = (blockIdx.x * 256 + threadIdx.x) * 8;
  if (i < n8 * 8) {
    v4f a = *(const v4f*)(in + i);
    v4f b = *(const v4f*)(in + i + 4);
    v8bf o;
    #pragma unroll
    for (int q = 0; q < 4; ++q) { o[q] = (__bf16)a[q]; o[4 + q] = (__bf16)b[q]; }
    *(v8bf*)(out + i) = o;
  }
}

// =====================================================================
// GEMM: C[M,N] = A[M,K] * B[N,K]^T + bias   (B is bf16, rows K-contiguous)
// 128x128 tile, BK=32, 256 threads = 8 waves, each wave 64x32 (8 WMMAs).
// Software pipeline: tile k+1 staged (regs or async-DMA) under tile k's
// WMMAs; ping-pong LDS buffers.
// =====================================================================
template<int N, int K, bool A_F32, bool OUT_BF16, bool CLAMP>
__global__ __launch_bounds__(256)
void gemm_wmma(const void* __restrict__ Ap, const __bf16* __restrict__ Bp,
               const float* __restrict__ bias, void* __restrict__ Cp)
{
  constexpr int KSTEPS = K / 32;
  // async loads issued per stage per thread (in-order completion):
  constexpr int NASYNC = HAVE_ASYNC_LDS ? ((A_F32 ? 0 : 2) + 2) : 0;

  // row stride 40 halves (80B): conflict-free 16-bank fragment reads
  __shared__ __bf16 As[2][128][40];
  __shared__ __bf16 Bs[2][128][40];

  const int tid  = threadIdx.x;
  const int lane = tid & 31;
  const int wave = tid >> 5;
  const int wm   = wave & 1;       // 64-row half of M tile
  const int wn   = wave >> 1;      // 32-col quarter of N tile
  const int lr   = lane & 15;
  const int kh   = lane >> 4;

  const int blockN = blockIdx.x * 128;
  const int blockM = blockIdx.y * 128;

  const int srow = tid >> 1;          // 0..127
  const int skk  = (tid & 1) * 16;    // 0 or 16

  const float*  Arowf = (const float*)Ap  + (size_t)(blockM + srow) * K + skk;
  const __bf16* Arowh = (const __bf16*)Ap + (size_t)(blockM + srow) * K + skk;
  const __bf16* Brow  = Bp                + (size_t)(blockN + srow) * K + skk;

  v8f zero = {};
  v8f acc[4][2];
  #pragma unroll
  for (int i = 0; i < 4; ++i)
    #pragma unroll
    for (int j = 0; j < 2; ++j)
      acc[i][j] = zero;

  v4f  afr[4];    // A f32 register stage
  v8bf ahr[2];    // A bf16 register stage (fallback only)
  v8bf bhr[2];    // B register stage (fallback only)

  // ---- stage helpers ----
  auto load_regs = [&](int k0) {
    if constexpr (A_F32) {
      afr[0] = *(const v4f*)(Arowf + k0);
      afr[1] = *(const v4f*)(Arowf + k0 + 4);
      afr[2] = *(const v4f*)(Arowf + k0 + 8);
      afr[3] = *(const v4f*)(Arowf + k0 + 12);
      __builtin_prefetch(Arowf + k0 + 64, 0, 1);   // 2 tiles ahead
    }
#if !HAVE_ASYNC_LDS
    if constexpr (!A_F32) {
      ahr[0] = *(const v8bf*)(Arowh + k0);
      ahr[1] = *(const v8bf*)(Arowh + k0 + 8);
    }
    bhr[0] = *(const v8bf*)(Brow + k0);
    bhr[1] = *(const v8bf*)(Brow + k0 + 8);
#endif
  };

  auto store_regs = [&](int bufi) {
    if constexpr (A_F32) {
      v8bf o0, o1;
      #pragma unroll
      for (int q = 0; q < 4; ++q) {
        o0[q]     = (__bf16)afr[0][q];
        o0[4 + q] = (__bf16)afr[1][q];
        o1[q]     = (__bf16)afr[2][q];
        o1[4 + q] = (__bf16)afr[3][q];
      }
      *(v8bf*)(&As[bufi][srow][skk])     = o0;
      *(v8bf*)(&As[bufi][srow][skk + 8]) = o1;
    }
#if !HAVE_ASYNC_LDS
    if constexpr (!A_F32) {
      *(v8bf*)(&As[bufi][srow][skk])     = ahr[0];
      *(v8bf*)(&As[bufi][srow][skk + 8]) = ahr[1];
    }
    *(v8bf*)(&Bs[bufi][srow][skk])     = bhr[0];
    *(v8bf*)(&Bs[bufi][srow][skk + 8]) = bhr[1];
#endif
  };

  auto issue_async = [&](int bufi, int k0) {
#if HAVE_ASYNC_LDS
    if constexpr (!A_F32) {
      async_cp_b128(Arowh + k0,     &As[bufi][srow][skk]);
      async_cp_b128(Arowh + k0 + 8, &As[bufi][srow][skk + 8]);
    }
    async_cp_b128(Brow + k0,     &Bs[bufi][srow][skk]);
    async_cp_b128(Brow + k0 + 8, &Bs[bufi][srow][skk + 8]);
#else
    (void)bufi; (void)k0;
#endif
  };

  // ---- pipeline prologue: tile 0 in flight ----
  load_regs(0);
  issue_async(0, 0);

  for (int kt = 0; kt < KSTEPS; ++kt) {
    const int buf = kt & 1;

    store_regs(buf);                       // regs(tile kt) -> LDS[buf]

    if (kt + 1 < KSTEPS) {                 // stage tile kt+1 under compute
      const int k0n = (kt + 1) << 5;
      load_regs(k0n);
      issue_async(buf ^ 1, k0n);
      wait_asynccnt<NASYNC>();             // tile kt's DMA done; kt+1 in flight
    } else {
      wait_asynccnt<0>();
    }
    __syncthreads();                       // LDS[buf] fully staged, all waves

    // ---- B fragments: lane = column, 16 contiguous K at kh*16 ----
    v16bf bfrag[2];
    #pragma unroll
    for (int j = 0; j < 2; ++j) {
      const int r = wn * 32 + j * 16 + lr;
      v8bf lo = *(const v8bf*)(&Bs[buf][r][kh * 16]);
      v8bf hi = *(const v8bf*)(&Bs[buf][r][kh * 16 + 8]);
      bfrag[j] = __builtin_shufflevector(lo, hi,
                   0,1,2,3,4,5,6,7,8,9,10,11,12,13,14,15);
    }
    // ---- A fragments + WMMA ----
    #pragma unroll
    for (int i = 0; i < 4; ++i) {
      const int r = wm * 64 + i * 16 + lr;
      v8bf lo = *(const v8bf*)(&As[buf][r][kh * 16]);       // K = kh*16+0..7
      v8bf hi = *(const v8bf*)(&As[buf][r][kh * 16 + 8]);   // K = kh*16+8..15
      v16bf afrag = __builtin_shufflevector(lo, hi,
                      0,1,2,3,4,5,6,7,8,9,10,11,12,13,14,15);
      #pragma unroll
      for (int j = 0; j < 2; ++j) {
        acc[i][j] = __builtin_amdgcn_wmma_f32_16x16x32_bf16(
            false, afrag, false, bfrag[j], (short)0, acc[i][j], false, false);
      }
    }
    __syncthreads();                       // allow restage of buf next iter
  }

  // ---- epilogue: bias, clamp, store ----
  #pragma unroll
  for (int i = 0; i < 4; ++i) {
    #pragma unroll
    for (int j = 0; j < 2; ++j) {
      const int col = blockN + wn * 32 + j * 16 + lr;
      const float bv = bias[col];
      #pragma unroll
      for (int r = 0; r < 8; ++r) {
        const int row = blockM + wm * 64 + i * 16 + kh * 8 + r;
        float v = acc[i][j][r] + bv;
        if constexpr (CLAMP) v = v < -1.f ? -1.f : (v > 1.f ? 1.f : v);
        if constexpr (OUT_BF16)
          ((__bf16*)Cp)[(size_t)row * N + col] = (__bf16)v;
        else
          ((float*)Cp)[(size_t)row * N + col] = v;
      }
    }
  }
}

// =====================================================================
// VQ middle: z = h*Wpi^T + bpi ; argmin_k ||z - e_k||^2 ; vq = e_idx*Wpo^T + bpo
// One token per wave (lane == codebook dim == 32).
// =====================================================================
__global__ __launch_bounds__(256)
void vq_mid(const __bf16* __restrict__ h,   const float* __restrict__ Wpi,
            const float* __restrict__ bpi,  const float* __restrict__ embed,
            const float* __restrict__ Wpo,  const float* __restrict__ bpo,
            __bf16* __restrict__ vqout, int* __restrict__ indices,
            float* __restrict__ loss)
{
  __shared__ float zsh[8][CODE_DIM];
  const int lane = threadIdx.x & 31;
  const int wave = threadIdx.x >> 5;
  const int t = blockIdx.x * 8 + wave;     // one token per wave

  if (blockIdx.x == 0 && threadIdx.x == 0) loss[0] = 0.f;  // commit loss == 0

  // ---- project_in: z[lane] ----
  const __bf16* hrow = h + (size_t)t * HIDDEN;      // broadcast across lanes
  const float*  wrow = Wpi + (size_t)lane * HIDDEN;
  float z = bpi[lane];
  for (int k = 0; k < HIDDEN; k += 8) {
    v8bf hv = *(const v8bf*)(hrow + k);
    v4f  w0 = *(const v4f*)(wrow + k);
    v4f  w1 = *(const v4f*)(wrow + k + 4);
    #pragma unroll
    for (int q = 0; q < 4; ++q) z += (float)hv[q] * w0[q];
    #pragma unroll
    for (int q = 0; q < 4; ++q) z += (float)hv[4 + q] * w1[q];
  }
  zsh[wave][lane] = z;
  __syncthreads();

  // ---- nearest code: each lane scores 8 of 256 codes ----
  float best = 3.4e38f;
  int   bidx = 0x7fffffff;
  for (int cc = 0; cc < 8; ++cc) {
    const int code = lane * 8 + cc;
    const float* e = embed + (size_t)code * CODE_DIM;
    float d = 0.f;
    #pragma unroll
    for (int c = 0; c < CODE_DIM; ++c) {
      const float diff = zsh[wave][c] - e[c];
      d += diff * diff;
    }
    if (d < best) { best = d; bidx = code; }
  }
  // wave32 argmin reduction, first-min (smallest index) tie-break
  #pragma unroll
  for (int off = 16; off > 0; off >>= 1) {
    const float od = __shfl_xor(best, off, 32);
    const int   oi = __shfl_xor(bidx, off, 32);
    if (od < best || (od == best && oi < bidx)) { best = od; bidx = oi; }
  }
  if (lane == 0) indices[t] = bidx;

  // ---- project_out from quantized code (straight-through forward value) ----
  const float* e = embed + (size_t)bidx * CODE_DIM;
  float ev[CODE_DIM];
  #pragma unroll
  for (int c = 0; c < CODE_DIM; ++c) ev[c] = e[c];
  #pragma unroll 4
  for (int j = 0; j < HIDDEN / 32; ++j) {
    const int oh = j * 32 + lane;
    const float* w = Wpo + (size_t)oh * CODE_DIM;
    float v = bpo[oh];
    #pragma unroll
    for (int c = 0; c < CODE_DIM; ++c) v += ev[c] * w[c];
    vqout[(size_t)t * HIDDEN + oh] = (__bf16)v;
  }
}

// =====================================================================
// launch
// =====================================================================
extern "C" void kernel_launch(void* const* d_in, const int* in_sizes, int n_in,
                              void* d_out, int out_size, void* d_ws, size_t ws_size,
                              hipStream_t stream)
{
  (void)in_sizes; (void)n_in; (void)out_size; (void)ws_size;

  const float* x     = (const float*)d_in[0];
  const float* Wdown = (const float*)d_in[1];
  const float* bdown = (const float*)d_in[2];
  const float* Wpi   = (const float*)d_in[3];
  const float* bpi   = (const float*)d_in[4];
  const float* embed = (const float*)d_in[5];
  const float* Wpo   = (const float*)d_in[6];
  const float* bpo   = (const float*)d_in[7];
  const float* Wup   = (const float*)d_in[8];
  const float* bup   = (const float*)d_in[9];

  // d_out = [ y : f32 32768*2048 | indices : i32 32768 | commit_loss : f32 1 ]
  float* y       = (float*)d_out;
  int*   indices = (int*)((char*)d_out + (size_t)TOKENS * OUT_FEAT * sizeof(float));
  float* loss    = (float*)(indices + TOKENS);

  // workspace: h bf16 (32MB) | vq bf16 (32MB) | Wdown bf16 (2MB) | Wup bf16 (2MB)
  char* ws = (char*)d_ws;
  __bf16* h_bf  = (__bf16*)(ws);
  __bf16* vq_bf = (__bf16*)(ws + (size_t)TOKENS * HIDDEN * 2);
  __bf16* Wd_bf = (__bf16*)(ws + (size_t)TOKENS * HIDDEN * 4);
  __bf16* Wu_bf = (__bf16*)(ws + (size_t)TOKENS * HIDDEN * 4 + (size_t)HIDDEN * IN_FEAT * 2);

  dim3 blk(256);

  // 0) pre-convert weights to bf16 (DMA-able by the async-LDS path)
  cvt_f32_to_bf16<<<dim3((HIDDEN * IN_FEAT / 8 + 255) / 256), blk, 0, stream>>>(
      Wdown, Wd_bf, HIDDEN * IN_FEAT / 8);
  cvt_f32_to_bf16<<<dim3((OUT_FEAT * HIDDEN / 8 + 255) / 256), blk, 0, stream>>>(
      Wup, Wu_bf, OUT_FEAT * HIDDEN / 8);

  // 1) h = x @ Wdown^T + bdown  -> bf16
  gemm_wmma<HIDDEN, IN_FEAT, true, true, false>
      <<<dim3(HIDDEN / 128, TOKENS / 128), blk, 0, stream>>>(
      x, Wd_bf, bdown, h_bf);

  // 2) VQ middle stage: z, argmin, project_out -> bf16 vq + indices + loss
  vq_mid<<<dim3(TOKENS / 8), blk, 0, stream>>>(
      h_bf, Wpi, bpi, embed, Wpo, bpo, vq_bf, indices, loss);

  // 3) y = clamp(vq @ Wup^T + bup, -1, 1) -> f32
  gemm_wmma<OUT_FEAT, HIDDEN, false, false, true>
      <<<dim3(OUT_FEAT / 128, TOKENS / 128), blk, 0, stream>>>(
      vq_bf, Wu_bf, bup, y);
}
